// DKRZSeismologyLSTM_44195213475891
// MI455X (gfx1250) — compile-verified, matrix operand order checked by reading, not run on hardware
//
#include <hip/hip_runtime.h>
#include <hip/hip_bf16.h>

// ---------------------------------------------------------------------------
// 2-layer LSTM (B=256, T=512, F=49, H=512) persistent kernel for gfx1250.
// Batch-parallel: 16 workgroups x 16 batch rows, no cross-WG communication.
// bf16 WMMA (v_wmma_f32_16x16x32_bf16) with f32 accumulation; h state in LDS
// (bf16); cell state in LDS (f32, lane-contiguous); weights pre-converted to
// bf16 in d_ws and streamed from L2 each step.
// 16 waves/WG x 8 gate tiles/wave; all tile/k offsets fold into load
// immediates (one address register per weight panel). Per k-block the 16
// weight loads are batched ahead of the 8 WMMAs via sched_group_barrier
// (VMEMx16 -> DSx2 -> WMMAx8), so only the first WMMA of a block waits on
// memory (14 loads still in flight) instead of every WMMA stalling.
// ---------------------------------------------------------------------------

typedef __attribute__((ext_vector_type(16))) __bf16 v16bf;
typedef __attribute__((ext_vector_type(8)))  __bf16 v8bf;
typedef __attribute__((ext_vector_type(8)))  float  v8f;

#define B_   256
#define T_   512
#define F_   49
#define H_   512
#define G_   2048     // 4*H
#define FP_  64       // F padded to multiple of 32 (K of one WMMA x2)
#define HSTR 528      // LDS row stride for h (bf16 elements), keeps 16B align
#define NTHR 512      // 16 waves

// sched_group_barrier masks
#define SGB_WMMA  0x008
#define SGB_VMRD  0x020
#define SGB_DSRD  0x100

// workspace layout (bytes)
#define OFF_WIH1 0
#define OFF_WHH1 (OFF_WIH1 + G_*FP_*2)
#define OFF_WIH2 (OFF_WHH1 + G_*H_*2)
#define OFF_WHH2 (OFF_WIH2 + G_*H_*2)
#define OFF_B1   (OFF_WHH2 + G_*H_*2)
#define OFF_B2   (OFF_B1   + G_*4)

__device__ __forceinline__ float sigf(float x) { return 1.f / (1.f + __expf(-x)); }
__device__ __forceinline__ float tanhfast(float x) { return 1.f - 2.f / (__expf(2.f*x) + 1.f); }

// Build a 16-element bf16 WMMA fragment: per ISA layout, each lane holds
// 8 contiguous K values and another 8 at +16 (lanes 16-31 offset by khi*8).
__device__ __forceinline__ v16bf frag16(const __bf16* p) {
  v8bf lo = *(const v8bf*)p;
  v8bf hi = *(const v8bf*)(p + 16);
  return __builtin_shufflevector(lo, hi, 0,1,2,3,4,5,6,7,8,9,10,11,12,13,14,15);
}

// Opaque pointer pass-through: breaks loop-invariant code motion so weight /
// bias loads are re-issued each timestep (L2-resident stream) instead of
// being hoisted out of the t-loop and spilled to scratch.
__device__ __forceinline__ const __bf16* opaque_bf(const __bf16* p) {
  unsigned long long u = (unsigned long long)p;
  asm volatile("" : "+s"(u));
  return (const __bf16*)u;
}
__device__ __forceinline__ const float* opaque_f(const float* p) {
  unsigned long long u = (unsigned long long)p;
  asm volatile("" : "+s"(u));
  return (const float*)u;
}

// One accumulation panel: acc[q][s] += A(16 x NKB*32) * W^T tiles.
// LDK is compile-time so every tile/k offset folds into the load immediate;
// the panel uses a single per-lane base address. Per k-block: batch all 16
// B loads, then the A DS reads, then the 8 WMMAs (pinned by group barriers).
template<int NKB, int LDK>
__device__ __forceinline__ void gemm_panel(v8f acc[4][2],
                                           const __bf16* w,
                                           const __bf16* a_base,
                                           const int astr,
                                           const int gbase, const int nlo,
                                           const int khi) {
  // single per-lane base: row (gbase+nlo), k-half khi
  const __bf16* wb = w + (size_t)(gbase + nlo) * LDK + khi * 8;
  const __bf16* ab = a_base + nlo * astr + khi * 8;

#pragma unroll
  for (int kb = 0; kb < NKB; ++kb) {
    v16bf bfrag[4][2];
#pragma unroll
    for (int q = 0; q < 4; ++q)
#pragma unroll
      for (int s = 0; s < 2; ++s)
        bfrag[q][s] = frag16(wb + (q * H_ + s * 16) * LDK + kb * 32);

    v16bf a = frag16(ab + kb * 32);

#pragma unroll
    for (int q = 0; q < 4; ++q)
#pragma unroll
      for (int s = 0; s < 2; ++s)
        acc[q][s] = __builtin_amdgcn_wmma_f32_16x16x32_bf16(
            false, a, false, bfrag[q][s], (short)0, acc[q][s], false, false);

    // Pin: 16 global B loads, 2 DS A loads, then 8 WMMAs for this k-block.
    __builtin_amdgcn_sched_group_barrier(SGB_VMRD, 16, 0);
    __builtin_amdgcn_sched_group_barrier(SGB_DSRD, 2, 0);
    __builtin_amdgcn_sched_group_barrier(SGB_WMMA, 8, 0);
  }
}

// ---------------------------------------------------------------------------
// Prep: convert weights f32 -> bf16 (pad w_ih1 K: 49 -> 64), fold biases.
// ---------------------------------------------------------------------------
__global__ void DKRZ_lstm_prep(const float* __restrict__ w_ih1,
                               const float* __restrict__ w_hh1,
                               const float* __restrict__ w_ih2,
                               const float* __restrict__ w_hh2,
                               const float* __restrict__ b_ih1,
                               const float* __restrict__ b_hh1,
                               const float* __restrict__ b_ih2,
                               const float* __restrict__ b_hh2,
                               unsigned short* __restrict__ wih1_u,
                               unsigned short* __restrict__ whh1_u,
                               unsigned short* __restrict__ wih2_u,
                               unsigned short* __restrict__ whh2_u,
                               float* __restrict__ b1,
                               float* __restrict__ b2) {
  __bf16* o_ih1 = (__bf16*)wih1_u;
  __bf16* o_hh1 = (__bf16*)whh1_u;
  __bf16* o_ih2 = (__bf16*)wih2_u;
  __bf16* o_hh2 = (__bf16*)whh2_u;
  const int N1 = G_ * FP_;        // padded w_ih1
  const int N2 = G_ * H_;         // each hidden matrix
  const int total = N1 + 3 * N2 + 2 * G_;
  for (int i = blockIdx.x * blockDim.x + threadIdx.x; i < total;
       i += gridDim.x * blockDim.x) {
    if (i < N1) {
      int g = i >> 6, f = i & 63;
      float v = (f < F_) ? w_ih1[g * F_ + f] : 0.f;
      o_ih1[i] = (__bf16)v;
    } else if (i < N1 + N2) {
      int j = i - N1;                     o_hh1[j] = (__bf16)w_hh1[j];
    } else if (i < N1 + 2 * N2) {
      int j = i - N1 - N2;                o_ih2[j] = (__bf16)w_ih2[j];
    } else if (i < N1 + 3 * N2) {
      int j = i - N1 - 2 * N2;            o_hh2[j] = (__bf16)w_hh2[j];
    } else if (i < N1 + 3 * N2 + G_) {
      int j = i - N1 - 3 * N2;            b1[j] = b_ih1[j] + b_hh1[j];
    } else {
      int j = i - N1 - 3 * N2 - G_;       b2[j] = b_ih2[j] + b_hh2[j];
    }
  }
}

// ---------------------------------------------------------------------------
// Persistent recurrence kernel. One WG = 16 batch rows, 16 waves (512 thr).
// Wave w owns hidden chunk [w*32, w*32+32): gate tiles q*512 + w*32 + {0,16}
// for q in {i,f,g,o}. i/f/g/o accumulators share (lane,vgpr) layout, so the
// LSTM cell is elementwise per accumulator register; c lives in LDS with a
// lane-contiguous layout (4 x ds_load_b128 / ds_store_b128 per layer).
// ---------------------------------------------------------------------------
__global__ __launch_bounds__(NTHR, 1)
void DKRZ_lstm2_persistent(const float* __restrict__ x,
                           const unsigned short* __restrict__ wih1_u,
                           const unsigned short* __restrict__ whh1_u,
                           const unsigned short* __restrict__ wih2_u,
                           const unsigned short* __restrict__ whh2_u,
                           const float* __restrict__ b1,
                           const float* __restrict__ b2,
                           const float* __restrict__ wout,
                           const float* __restrict__ bout,
                           float* __restrict__ out) {
  __shared__ __align__(16) __bf16 x_lds[16 * FP_];
  __shared__ __align__(16) __bf16 h1_lds[16 * HSTR];
  __shared__ __align__(16) __bf16 h2_lds[16 * HSTR];
  __shared__ __align__(16) float  c_lds[2 * NTHR * 16];   // 64 KB, lane-major

  const int tid  = threadIdx.x;
  const int wave = tid >> 5;        // 0..15  -> hidden chunk w*32
  const int lane = tid & 31;
  const int nlo  = lane & 15;       // column within 16x16 tile
  const int khi  = lane >> 4;       // K half select for A/B fragments
  const int gb   = wave * 32;       // hidden base for this wave
  const int row0 = blockIdx.x * 16; // batch tile base

  float* cme1 = &c_lds[tid * 16];              // this lane's c1 (16 f32)
  float* cme2 = &c_lds[NTHR * 16 + tid * 16];  // this lane's c2 (16 f32)

  // zero-init recurrent state in LDS
  for (int i = tid; i < 16 * HSTR; i += NTHR) {
    h1_lds[i] = (__bf16)0.f;
    h2_lds[i] = (__bf16)0.f;
  }
  for (int i = tid; i < 2 * NTHR * 16; i += NTHR) c_lds[i] = 0.f;

  const float bo = bout[0];

  for (int t = 0; t < T_; ++t) {
    // fresh opaque bases each step: loads cannot be hoisted/spilled
    const __bf16* wih1 = opaque_bf((const __bf16*)wih1_u);
    const __bf16* whh1 = opaque_bf((const __bf16*)whh1_u);
    const __bf16* wih2 = opaque_bf((const __bf16*)wih2_u);
    const __bf16* whh2 = opaque_bf((const __bf16*)whh2_u);
    const float*  pb1  = opaque_f(b1);
    const float*  pb2  = opaque_f(b2);

    __syncthreads();
    // stage x[:,t,:] tile: 16x49 f32 -> 16x64 bf16 (zero padded K)
    for (int i = tid; i < 16 * FP_; i += NTHR) {
      int m = i >> 6, f = i & 63;
      float v = (f < F_) ? x[((size_t)(row0 + m) * T_ + t) * F_ + f] : 0.f;
      x_lds[m * FP_ + f] = (__bf16)v;
    }
    __syncthreads();

    // ---------------- layer 1: gates = x@Wih1^T + h1@Whh1^T + bias ----------
    v8f acc[4][2];
#pragma unroll
    for (int q = 0; q < 4; ++q)
#pragma unroll
      for (int s = 0; s < 2; ++s) {
        float bv = pb1[q * H_ + gb + s * 16 + nlo];
        acc[q][s] = (v8f){bv, bv, bv, bv, bv, bv, bv, bv};
      }
    gemm_panel<2, FP_>(acc, wih1, x_lds, FP_, gb, nlo, khi);   // x part
    gemm_panel<16, H_>(acc, whh1, h1_lds, HSTR, gb, nlo, khi); // h1 part
    __syncthreads();  // all reads of old h1 complete across waves

    // LSTM cell 1 — elementwise on accumulators; c1 round-trips LDS
#pragma unroll
    for (int s = 0; s < 2; ++s)
#pragma unroll
      for (int r = 0; r < 8; ++r) {
        float iv = acc[0][s][r], fv = acc[1][s][r];
        float gv = acc[2][s][r], ov = acc[3][s][r];
        float cn = sigf(fv) * cme1[s * 8 + r] + sigf(iv) * tanhfast(gv);
        cme1[s * 8 + r] = cn;
        float hn = sigf(ov) * tanhfast(cn);
        int m = r + 8 * khi;
        h1_lds[m * HSTR + gb + s * 16 + nlo] = (__bf16)hn;
      }
    __syncthreads();

    // ---------------- layer 2: gates = h1@Wih2^T + h2@Whh2^T + bias ---------
#pragma unroll
    for (int q = 0; q < 4; ++q)
#pragma unroll
      for (int s = 0; s < 2; ++s) {
        float bv = pb2[q * H_ + gb + s * 16 + nlo];
        acc[q][s] = (v8f){bv, bv, bv, bv, bv, bv, bv, bv};
      }
    gemm_panel<16, H_>(acc, wih2, h1_lds, HSTR, gb, nlo, khi); // h1(new)
    gemm_panel<16, H_>(acc, whh2, h2_lds, HSTR, gb, nlo, khi); // h2(old)
    __syncthreads();  // all reads of old h2 complete

    // LSTM cell 2
#pragma unroll
    for (int s = 0; s < 2; ++s)
#pragma unroll
      for (int r = 0; r < 8; ++r) {
        float iv = acc[0][s][r], fv = acc[1][s][r];
        float gv = acc[2][s][r], ov = acc[3][s][r];
        float cn = sigf(fv) * cme2[s * 8 + r] + sigf(iv) * tanhfast(gv);
        cme2[s * 8 + r] = cn;
        float hn = sigf(ov) * tanhfast(cn);
        int m = r + 8 * khi;
        h2_lds[m * HSTR + gb + s * 16 + nlo] = (__bf16)hn;
      }
    __syncthreads();

    // output projection: wave w reduces batch row w of h2 against w_out
    {
      const float* wo = opaque_f(wout);
      float sum = 0.f;
      const __bf16* hrow = &h2_lds[wave * HSTR];
#pragma unroll 4
      for (int c = lane; c < H_; c += 32) sum += (float)hrow[c] * wo[c];
#pragma unroll
      for (int off = 16; off >= 1; off >>= 1) sum += __shfl_down(sum, off, 32);
      if (lane == 0) out[(size_t)(row0 + wave) * T_ + t] = sum + bo;
    }
  }
}

// ---------------------------------------------------------------------------
extern "C" void kernel_launch(void* const* d_in, const int* in_sizes, int n_in,
                              void* d_out, int out_size, void* d_ws, size_t ws_size,
                              hipStream_t stream) {
  const float* x     = (const float*)d_in[0];
  const float* w_ih1 = (const float*)d_in[1];
  const float* w_hh1 = (const float*)d_in[2];
  const float* b_ih1 = (const float*)d_in[3];
  const float* b_hh1 = (const float*)d_in[4];
  const float* w_ih2 = (const float*)d_in[5];
  const float* w_hh2 = (const float*)d_in[6];
  const float* b_ih2 = (const float*)d_in[7];
  const float* b_hh2 = (const float*)d_in[8];
  const float* w_out = (const float*)d_in[9];
  const float* b_out = (const float*)d_in[10];
  float* out = (float*)d_out;

  char* ws = (char*)d_ws;
  unsigned short* wih1p = (unsigned short*)(ws + OFF_WIH1);
  unsigned short* whh1p = (unsigned short*)(ws + OFF_WHH1);
  unsigned short* wih2p = (unsigned short*)(ws + OFF_WIH2);
  unsigned short* whh2p = (unsigned short*)(ws + OFF_WHH2);
  float* b1 = (float*)(ws + OFF_B1);
  float* b2 = (float*)(ws + OFF_B2);

  DKRZ_lstm_prep<<<2048, 512, 0, stream>>>(
      w_ih1, w_hh1, w_ih2, w_hh2, b_ih1, b_hh1, b_ih2, b_hh2,
      wih1p, whh1p, wih2p, whh2p, b1, b2);

  DKRZ_lstm2_persistent<<<B_ / 16, NTHR, 0, stream>>>(
      x, wih1p, whh1p, wih2p, whh2p, b1, b2, w_out, b_out, out);
}
